// SwinTransformerBlock_14285061226732
// MI455X (gfx1250) — matrix operand order repaired and checked by reading, not compiled, not run on hardware
//
#include <hip/hip_runtime.h>
#include <hip/hip_bf16.h>
#include <math.h>

typedef __attribute__((ext_vector_type(16))) _Float16 v16h;
typedef __attribute__((ext_vector_type(8)))  _Float16 v8h;
typedef __attribute__((ext_vector_type(8)))  float    v8f;

#define WSZ   7
#define SHF   3
#define DIMC  96
#define NHD   3
#define HDM   32
#define NBAT  4
#define SSZ   28
#define NWIN  256     // NBAT * 64 windows
#define LTOK  343     // 7^3 tokens per window
#define LPAD  352     // 22 * 16
#define TOKS  87808   // NBAT * 28^3  (== 5488 * 16)
#define RPBT  2197    // 13^3

// ---------------- WMMA fragment helpers (gfx1250 16x16x32 f16) ----------------
// A fragment: 16(M) x 32(K), row-major source, row stride lda (f16 elems).
//   half = lane>>4 : elems 0..7 -> K = half*8 + 0..7 ; elems 8..15 -> K = 16 + half*8 + 0..7
__device__ __forceinline__ v16h frag_a(const _Float16* p, int lda) {
  const int lane = threadIdx.x & 31;
  const _Float16* r = p + (lane & 15) * lda + (lane >> 4) * 8;
  v16h a;
#pragma unroll
  for (int g = 0; g < 8; ++g) {
    const int kb = (g < 4 ? 0 : 16) + (g & 3) * 2;
    a[2 * g]     = r[kb];
    a[2 * g + 1] = r[kb + 1];
  }
  return a;
}

// B fragment from BT: N(16 cols) x K(32), BT row-major (n, k), row stride ldb.
// lane col = lane&15 ; elems 0..7 -> K = half*8 + 0..7 ; elems 8..15 -> K = 16 + half*8 + 0..7
__device__ __forceinline__ v16h frag_b(const _Float16* p, int ldb) {
  const int lane = threadIdx.x & 31;
  const _Float16* r = p + (lane & 15) * ldb + (lane >> 4) * 8;
  v16h b;
#pragma unroll
  for (int e = 0; e < 16; ++e) b[e] = r[e < 8 ? e : 8 + e];
  return b;
}

__device__ __forceinline__ v8f wmma_f16(v16h a, v16h b, v8f c) {
  return __builtin_amdgcn_wmma_f32_16x16x32_f16(false, a, false, b, (short)0, c,
                                                false, false);
}

// ---------------- K0: f32 -> f16 weight convert ----------------
__global__ void k_cvt(const float* __restrict__ s, _Float16* __restrict__ d, int n) {
  int i = blockIdx.x * blockDim.x + threadIdx.x;
  if (i < n) d[i] = (_Float16)s[i];
}

// ---------------- K1: fused (mask ? rel-pos-bias : -inf) f16 table ------------
// layout: tab[head][window][m][j], m,j < LTOK
__global__ void k_masktab(const float* __restrict__ rpb, const int* __restrict__ rpi,
                          const unsigned char* __restrict__ amask,
                          _Float16* __restrict__ tab) {
  const size_t total = (size_t)NHD * 64 * LTOK * LTOK;
  size_t i = (size_t)blockIdx.x * blockDim.x + threadIdx.x;
  if (i >= total) return;
  const size_t LL2 = (size_t)LTOK * LTOK;
  const int h = (int)(i / (64 * LL2));
  const size_t r = i - (size_t)h * 64 * LL2;
  const int w = (int)(r / LL2);
  const int mj = (int)(r - (size_t)w * LL2);
  float v = -INFINITY;
  if (amask[(size_t)w * LL2 + mj]) v = rpb[h * RPBT + rpi[mj]];
  tab[i] = (_Float16)v;
}

// ---------------- K2: LN1 + shift-roll + window partition -> f16 --------------
__global__ void k_ln1_part(const float* __restrict__ x, const float* __restrict__ w,
                           const float* __restrict__ b, _Float16* __restrict__ h16) {
  const int blk = blockIdx.x;          // win * LPAD + l
  const int win = blk / LPAD;
  const int l   = blk - win * LPAD;
  const int c   = threadIdx.x;         // 0..95
  if (l >= LTOK) { h16[(size_t)blk * DIMC + c] = (_Float16)0.f; return; }
  const int bi = win >> 6, wi = win & 63;
  const int g0 = wi >> 4, g1 = (wi >> 2) & 3, g2 = wi & 3;
  const int i0 = l / 49, rm = l - i0 * 49, i1 = rm / 7, i2 = rm - i1 * 7;
  int p0 = g0 * WSZ + i0 - SHF; if (p0 < 0) p0 += SSZ;
  int p1 = g1 * WSZ + i1 - SHF; if (p1 < 0) p1 += SSZ;
  int p2 = g2 * WSZ + i2 - SHF; if (p2 < 0) p2 += SSZ;
  const size_t src = ((((size_t)bi * SSZ + p0) * SSZ + p1) * SSZ + p2) * DIMC;
  float v = x[src + c];
  __shared__ float part[3];
  float s = v;
#pragma unroll
  for (int o = 16; o; o >>= 1) s += __shfl_xor(s, o, 32);
  if ((threadIdx.x & 31) == 0) part[threadIdx.x >> 5] = s;
  __syncthreads();
  const float mean = (part[0] + part[1] + part[2]) * (1.f / 96.f);
  __syncthreads();
  const float dv = v - mean;
  float q = dv * dv;
#pragma unroll
  for (int o = 16; o; o >>= 1) q += __shfl_xor(q, o, 32);
  if ((threadIdx.x & 31) == 0) part[threadIdx.x >> 5] = q;
  __syncthreads();
  const float var = (part[0] + part[1] + part[2]) * (1.f / 96.f);
  const float y = dv * rsqrtf(var + 1e-5f) * w[c] + b[c];
  h16[(size_t)blk * DIMC + c] = (_Float16)y;
}

// ---------------- K3: QKV GEMM (h @ Wqkv^T + b), q scaled, v transposed -------
// 16-wide N tiles never straddle q/k/v or head boundaries -> routing is
// wave-uniform: t = nt/6, head = (nt%6)>>1, d = (nt&1)*16 + col.
__global__ void k_qkv(const _Float16* __restrict__ h16, const _Float16* __restrict__ wq,
                      const float* __restrict__ bq, _Float16* __restrict__ q16,
                      _Float16* __restrict__ k16, _Float16* __restrict__ vT16) {
  const int wave = (int)((blockIdx.x * blockDim.x + threadIdx.x) >> 5);
  const int nTiles = NWIN * 22 * 18;
  if (wave >= nTiles) return;
  const int win = wave / (22 * 18);
  int r = wave - win * 22 * 18;
  const int mt = r / 18, nt = r - mt * 18;
  const _Float16* ab = h16 + ((size_t)win * LPAD + mt * 16) * DIMC;
  const _Float16* bb = wq + (size_t)(nt * 16) * DIMC;
  v8f c = {};
#pragma unroll
  for (int kk = 0; kk < 3; ++kk)
    c = wmma_f16(frag_a(ab + kk * 32, DIMC), frag_b(bb + kk * 32, DIMC), c);
  const int lane = threadIdx.x & 31;
  const int col = lane & 15, half = lane >> 4;
  const int t    = nt / 6;            // 0=q 1=k 2=v  (wave-uniform)
  const int head = (nt % 6) >> 1;     // wave-uniform
  const int d    = (nt & 1) * 16 + col;
  const float bias = bq[nt * 16 + col];
  const float scale = 0.1767766952966369f;  // 32^-0.5
  if (t == 2) {
    // vT[d][m]: 8 consecutive m per lane -> one aligned 16B store
    v8h pk;
#pragma unroll
    for (int rr = 0; rr < 8; ++rr) pk[rr] = (_Float16)(c[rr] + bias);
    _Float16* dst = vT16 + (((size_t)win * NHD + head) * HDM + d) * LPAD +
                    mt * 16 + half * 8;
    *(v8h*)dst = pk;
  } else {
    _Float16* dst = (t == 0 ? q16 : k16) +
                    (((size_t)win * NHD + head) * LPAD + mt * 16) * HDM + d;
    const float sc = (t == 0) ? scale : 1.f;
#pragma unroll
    for (int rr = 0; rr < 8; ++rr)
      dst[(size_t)(half * 8 + rr) * HDM] = (_Float16)((c[rr] + bias) * sc);
  }
}

// ---------------- K4: attention (scores + fused bias/mask + softmax + P@V) ----
__global__ void __launch_bounds__(128)
k_attn(const _Float16* __restrict__ q16, const _Float16* __restrict__ k16,
       const _Float16* __restrict__ vT16, const _Float16* __restrict__ mtab,
       _Float16* __restrict__ o16) {
  __shared__ float    sS[16][LPAD];
  __shared__ _Float16 sP[16][LPAD];
  const int rt   = blockIdx.x;   // row tile 0..21
  const int head = blockIdx.y;   // 0..2
  const int win  = blockIdx.z;   // 0..255
  const int wmod = win & 63;
  const int wave = threadIdx.x >> 5;
  const int lane = threadIdx.x & 31;
  const int col = lane & 15, half = lane >> 4;
  const _Float16* qb = q16 + (((size_t)win * NHD + head) * LPAD + rt * 16) * HDM;
  const _Float16* kb = k16 + (((size_t)win * NHD + head) * LPAD) * HDM;
  const _Float16* tb = mtab + ((size_t)head * 64 + wmod) * LTOK * LTOK;

  const v16h a0 = frag_a(qb, HDM);           // 16 rows x full K=32 (head dim)
  for (int ct = wave; ct < 22; ct += 4) {    // score strip 16 x 352
    v8f c = {};
    c = wmma_f16(a0, frag_b(kb + (size_t)(ct * 16) * HDM, HDM), c);
    const int j = ct * 16 + col;
#pragma unroll
    for (int rr = 0; rr < 8; ++rr) {
      const int lr = half * 8 + rr;
      const int m = rt * 16 + lr;
      float s = -3.0e38f;
      if (m < LTOK && j < LTOK)
        s = c[rr] + (float)tb[(size_t)m * LTOK + j];   // -inf where masked
      sS[lr][j] = s;
    }
  }
  __syncthreads();
  // row softmax (one wave per row, 4 rows per wave)
  for (int lr = wave; lr < 16; lr += 4) {
    const int m = rt * 16 + lr;
    if (m >= LTOK) {
      for (int jj = lane; jj < LPAD; jj += 32) sP[lr][jj] = (_Float16)0.f;
      continue;
    }
    float mx = -3.0e38f;
    for (int jj = lane; jj < LPAD; jj += 32) mx = fmaxf(mx, sS[lr][jj]);
#pragma unroll
    for (int o = 16; o; o >>= 1) mx = fmaxf(mx, __shfl_xor(mx, o, 32));
    float sum = 0.f;
    for (int jj = lane; jj < LPAD; jj += 32) {
      const float e = __expf(sS[lr][jj] - mx);
      sum += e;
      sS[lr][jj] = e;
    }
#pragma unroll
    for (int o = 16; o; o >>= 1) sum += __shfl_xor(sum, o, 32);
    const float inv = 1.f / sum;
    for (int jj = lane; jj < LPAD; jj += 32) sP[lr][jj] = (_Float16)(sS[lr][jj] * inv);
  }
  __syncthreads();
  // O = P @ V : two 16x16 output tiles (d = 0..15, 16..31), waves 0 and 1
  if (wave < 2) {
    const int dt = wave;
    const _Float16* vb = vT16 + (((size_t)win * NHD + head) * HDM + dt * 16) * LPAD;
    v8f c = {};
    for (int kt = 0; kt < 11; ++kt)   // K = 352 = 11 * 32
      c = wmma_f16(frag_a(&sP[0][kt * 32], LPAD), frag_b(vb + kt * 32, LPAD), c);
    const int d = dt * 16 + col;
#pragma unroll
    for (int rr = 0; rr < 8; ++rr) {
      const int m = rt * 16 + half * 8 + rr;
      o16[((size_t)win * LPAD + m) * DIMC + head * HDM + d] = (_Float16)c[rr];
    }
  }
}

// ---------------- K5: proj GEMM + window reverse + roll back + residual -------
__global__ void k_proj(const _Float16* __restrict__ o16, const _Float16* __restrict__ wp,
                       const float* __restrict__ bp, const float* __restrict__ x,
                       float* __restrict__ xout) {
  const int wave = (int)((blockIdx.x * blockDim.x + threadIdx.x) >> 5);
  const int nTiles = NWIN * 22 * 6;
  if (wave >= nTiles) return;
  const int win = wave / 132;
  int r = wave - win * 132;
  const int mt = r / 6, nt = r - mt * 6;
  const _Float16* ab = o16 + ((size_t)win * LPAD + mt * 16) * DIMC;
  const _Float16* bb = wp + (size_t)(nt * 16) * DIMC;
  v8f c = {};
#pragma unroll
  for (int kk = 0; kk < 3; ++kk)
    c = wmma_f16(frag_a(ab + kk * 32, DIMC), frag_b(bb + kk * 32, DIMC), c);
  const int lane = threadIdx.x & 31;
  const int col = lane & 15, half = lane >> 4;
  const int n = nt * 16 + col;
  const float bias = bp[n];
  const int bi = win >> 6, wi = win & 63;
  const int g0 = wi >> 4, g1 = (wi >> 2) & 3, g2 = wi & 3;
#pragma unroll
  for (int rr = 0; rr < 8; ++rr) {
    const int m = mt * 16 + half * 8 + rr;
    if (m >= LTOK) continue;
    const int i0 = m / 49, rm = m - i0 * 49, i1 = rm / 7, i2 = rm - i1 * 7;
    int p0 = g0 * WSZ + i0 - SHF; if (p0 < 0) p0 += SSZ;
    int p1 = g1 * WSZ + i1 - SHF; if (p1 < 0) p1 += SSZ;
    int p2 = g2 * WSZ + i2 - SHF; if (p2 < 0) p2 += SSZ;
    const size_t idx = ((((size_t)bi * SSZ + p0) * SSZ + p1) * SSZ + p2) * DIMC + n;
    xout[idx] = x[idx] + c[rr] + bias;
  }
}

// ---------------- K6: LN2 -> f16 ----------------
__global__ void k_ln2(const float* __restrict__ xo, const float* __restrict__ w,
                      const float* __restrict__ b, _Float16* __restrict__ h) {
  const int tkn = blockIdx.x;
  const int c = threadIdx.x;
  const float v = xo[(size_t)tkn * DIMC + c];
  __shared__ float part[3];
  float s = v;
#pragma unroll
  for (int o = 16; o; o >>= 1) s += __shfl_xor(s, o, 32);
  if ((threadIdx.x & 31) == 0) part[threadIdx.x >> 5] = s;
  __syncthreads();
  const float mean = (part[0] + part[1] + part[2]) * (1.f / 96.f);
  __syncthreads();
  const float dv = v - mean;
  float q = dv * dv;
#pragma unroll
  for (int o = 16; o; o >>= 1) q += __shfl_xor(q, o, 32);
  if ((threadIdx.x & 31) == 0) part[threadIdx.x >> 5] = q;
  __syncthreads();
  const float var = (part[0] + part[1] + part[2]) * (1.f / 96.f);
  h[(size_t)tkn * DIMC + c] = (_Float16)(dv * rsqrtf(var + 1e-5f) * w[c] + b[c]);
}

// ---------------- K7: MLP fc1 + exact GELU ----------------
__global__ void k_mlp1(const _Float16* __restrict__ h, const _Float16* __restrict__ w1,
                       const float* __restrict__ b1, _Float16* __restrict__ mid) {
  const int wave = (int)((blockIdx.x * blockDim.x + threadIdx.x) >> 5);
  const int nTiles = (TOKS / 16) * 24;
  if (wave >= nTiles) return;
  const int mt = wave / 24, nt = wave - mt * 24;
  const _Float16* ab = h + (size_t)(mt * 16) * DIMC;
  const _Float16* bb = w1 + (size_t)(nt * 16) * DIMC;
  v8f c = {};
#pragma unroll
  for (int kk = 0; kk < 3; ++kk)
    c = wmma_f16(frag_a(ab + kk * 32, DIMC), frag_b(bb + kk * 32, DIMC), c);
  const int lane = threadIdx.x & 31;
  const int col = lane & 15, half = lane >> 4;
  const int n = nt * 16 + col;
  const float bias = b1[n];
#pragma unroll
  for (int rr = 0; rr < 8; ++rr) {
    const int m = mt * 16 + half * 8 + rr;
    const float z = c[rr] + bias;
    const float g = 0.5f * z * (1.f + erff(z * 0.70710678118654752f));
    mid[(size_t)m * 384 + n] = (_Float16)g;
  }
}

// ---------------- K8: MLP fc2 + final residual ----------------
__global__ void k_mlp2(const _Float16* __restrict__ mid, const _Float16* __restrict__ w2,
                       const float* __restrict__ b2, const float* __restrict__ xo,
                       float* __restrict__ out) {
  const int wave = (int)((blockIdx.x * blockDim.x + threadIdx.x) >> 5);
  const int nTiles = (TOKS / 16) * 6;
  if (wave >= nTiles) return;
  const int mt = wave / 6, nt = wave - mt * 6;
  const _Float16* ab = mid + (size_t)(mt * 16) * 384;
  const _Float16* bb = w2 + (size_t)(nt * 16) * 384;
  v8f c = {};
#pragma unroll
  for (int kk = 0; kk < 12; ++kk)
    c = wmma_f16(frag_a(ab + kk * 32, 384), frag_b(bb + kk * 32, 384), c);
  const int lane = threadIdx.x & 31;
  const int col = lane & 15, half = lane >> 4;
  const int n = nt * 16 + col;
  const float bias = b2[n];
#pragma unroll
  for (int rr = 0; rr < 8; ++rr) {
    const int m = mt * 16 + half * 8 + rr;
    const size_t idx = (size_t)m * DIMC + n;
    out[idx] = xo[idx] + c[rr] + bias;
  }
}

// ------------------------------- launcher ------------------------------------
extern "C" void kernel_launch(void* const* d_in, const int* in_sizes, int n_in,
                              void* d_out, int out_size, void* d_ws, size_t ws_size,
                              hipStream_t stream) {
  const float* x      = (const float*)d_in[0];
  const float* qkv_w  = (const float*)d_in[1];
  const float* qkv_b  = (const float*)d_in[2];
  const float* proj_w = (const float*)d_in[3];
  const float* proj_b = (const float*)d_in[4];
  const float* rpb    = (const float*)d_in[5];
  const float* ln1_w  = (const float*)d_in[6];
  const float* ln1_b  = (const float*)d_in[7];
  const float* ln2_w  = (const float*)d_in[8];
  const float* ln2_b  = (const float*)d_in[9];
  const float* mlp_w1 = (const float*)d_in[10];
  const float* mlp_b1 = (const float*)d_in[11];
  const float* mlp_w2 = (const float*)d_in[12];
  const float* mlp_b2 = (const float*)d_in[13];
  const unsigned char* amask = (const unsigned char*)d_in[14];
  const int* rpi      = (const int*)d_in[15];
  float* outp = (float*)d_out;

  char* base = (char*)d_ws;
  size_t off = 0;
  auto alloc = [&](size_t bytes) -> void* {
    void* r = base + off;
    off = (off + bytes + 255) & ~(size_t)255;
    return r;
  };
  _Float16* wQ   = (_Float16*)alloc((size_t)288 * 96 * 2);
  _Float16* wP   = (_Float16*)alloc((size_t)96 * 96 * 2);
  _Float16* wM1  = (_Float16*)alloc((size_t)384 * 96 * 2);
  _Float16* wM2  = (_Float16*)alloc((size_t)96 * 384 * 2);
  _Float16* mtab = (_Float16*)alloc((size_t)NHD * 64 * LTOK * LTOK * 2);
  _Float16* h16  = (_Float16*)alloc((size_t)NWIN * LPAD * DIMC * 2);  // reused as O16
  _Float16* q16  = (_Float16*)alloc((size_t)NWIN * NHD * LPAD * HDM * 2);
  _Float16* k16  = (_Float16*)alloc((size_t)NWIN * NHD * LPAD * HDM * 2);
  _Float16* vT16 = (_Float16*)alloc((size_t)NWIN * NHD * HDM * LPAD * 2);
  float*    xo   = (float*)   alloc((size_t)TOKS * DIMC * 4);
  _Float16* l2h  = (_Float16*)alloc((size_t)TOKS * DIMC * 2);
  _Float16* mid  = (_Float16*)alloc((size_t)TOKS * 384 * 2);
  _Float16* o16  = h16;  // reuse: h16 dead after QKV GEMM

  // K0: weight converts
  k_cvt<<<(288 * 96 + 255) / 256, 256, 0, stream>>>(qkv_w, wQ, 288 * 96);
  k_cvt<<<(96 * 96 + 255) / 256, 256, 0, stream>>>(proj_w, wP, 96 * 96);
  k_cvt<<<(384 * 96 + 255) / 256, 256, 0, stream>>>(mlp_w1, wM1, 384 * 96);
  k_cvt<<<(96 * 384 + 255) / 256, 256, 0, stream>>>(mlp_w2, wM2, 96 * 384);
  // K1: fused mask+bias table
  {
    size_t total = (size_t)NHD * 64 * LTOK * LTOK;
    k_masktab<<<(unsigned)((total + 255) / 256), 256, 0, stream>>>(rpb, rpi, amask, mtab);
  }
  // K2: LN1 + shift + window partition
  k_ln1_part<<<NWIN * LPAD, 96, 0, stream>>>(x, ln1_w, ln1_b, h16);
  // K3: QKV GEMM
  {
    int waves = NWIN * 22 * 18;
    k_qkv<<<(waves * 32 + 255) / 256, 256, 0, stream>>>(h16, wQ, qkv_b, q16, k16, vT16);
  }
  // K4: attention
  k_attn<<<dim3(22, NHD, NWIN), 128, 0, stream>>>(q16, k16, vT16, mtab, o16);
  // K5: proj + window reverse + roll back + residual
  {
    int waves = NWIN * 22 * 6;
    k_proj<<<(waves * 32 + 255) / 256, 256, 0, stream>>>(o16, wP, proj_b, x, xo);
  }
  // K6: LN2
  k_ln2<<<TOKS, 96, 0, stream>>>(xo, ln2_w, ln2_b, l2h);
  // K7: MLP fc1 + GELU
  {
    int waves = (TOKS / 16) * 24;
    k_mlp1<<<(waves * 32 + 255) / 256, 256, 0, stream>>>(l2h, wM1, mlp_b1, mid);
  }
  // K8: MLP fc2 + final residual
  {
    int waves = (TOKS / 16) * 6;
    k_mlp2<<<(waves * 32 + 255) / 256, 256, 0, stream>>>(mid, wM2, mlp_b2, xo, outp);
  }
  (void)in_sizes; (void)n_in; (void)out_size; (void)ws_size;
}
